// ExtraBlock_89696097009755
// MI455X (gfx1250) — matrix operand order repaired and checked by reading, not compiled, not run on hardware
//
#include <hip/hip_runtime.h>
#include <hip/hip_bf16.h>

typedef __attribute__((ext_vector_type(16))) __bf16 v16bf;
typedef __attribute__((ext_vector_type(8)))  __bf16 v8bf;
typedef __attribute__((ext_vector_type(4)))  __bf16 v4bf;
typedef __attribute__((ext_vector_type(8)))  float  v8f;

// Problem constants (match reference setup_inputs)
constexpr int N_IN   = 200000;
constexpr int N_OUT  = 25000;
constexpr int KOFF   = 27;
constexpr int MPAIR  = 200000;
constexpr int C_IN   = 128;
constexpr int C_OUT  = 512;
constexpr int NBATCH = 8;
constexpr float EPS  = 1e-6f;

constexpr int MT = 32;                       // gathered rows per block tile
constexpr int TILES_PER_K = MPAIR / MT;      // 6250 (exact)
constexpr int APAD = 8;                      // LDS row padding (bf16 elems) to rotate banks

// ---------------- prep: f32 -> bf16 feats ----------------
__global__ void cvt_feats_kernel(const float* __restrict__ f,
                                 __bf16* __restrict__ o, int n4) {
    int i = blockIdx.x * blockDim.x + threadIdx.x;
    if (i >= n4) return;
    float4 v = ((const float4*)f)[i];
    v4bf r;
    r.x = (__bf16)v.x; r.y = (__bf16)v.y; r.z = (__bf16)v.z; r.w = (__bf16)v.w;
    ((v4bf*)o)[i] = r;
}

// ---------------- prep: weights f32 [K][C_IN][C_OUT] -> bf16 panels -------
// Panel layout: wp[((k*32 + n/16)*16 + n%16)*C_IN + kk]  (lane-contiguous K)
__global__ void cvt_weights_kernel(const float* __restrict__ w,
                                   __bf16* __restrict__ wp, int total) {
    int i = blockIdx.x * blockDim.x + threadIdx.x;
    if (i >= total) return;
    int n  = i % C_OUT;
    int kk = (i / C_OUT) % C_IN;
    int k  = i / (C_OUT * C_IN);
    size_t dst = ((size_t)(k * 32 + (n >> 4)) * 16 + (n & 15)) * C_IN + kk;
    wp[dst] = (__bf16)w[i];
}

// ---------------- sparse conv: gather -> WMMA GEMM -> atomic scatter ------
__global__ __launch_bounds__(256)
void conv_wmma_kernel(const __bf16* __restrict__ featsBf,
                      const __bf16* __restrict__ wpanel,
                      const int*    __restrict__ in_idx,
                      const int*    __restrict__ out_idx,
                      float*        __restrict__ y) {
    __shared__ alignas(16) __bf16 Ash[MT][C_IN + APAD];
    __shared__ int iidx_sh[MT];
    __shared__ int oidx_sh[MT];

    const int bx    = blockIdx.x;
    const int k     = bx / TILES_PER_K;
    const int tile  = bx % TILES_PER_K;
    const int mbase = tile * MT;
    const int tid   = threadIdx.x;

    if (tid < MT) {
        iidx_sh[tid] = in_idx[(size_t)k * MPAIR + mbase + tid];
        oidx_sh[tid] = out_idx[(size_t)k * MPAIR + mbase + tid];
    }
    __syncthreads();

    // Stage gathered A tile (32 rows x 128 bf16) straight into LDS via the
    // CDNA5 async global->LDS path (no VGPR round-trip, tracked by ASYNCcnt).
    // 512 chunks of 16B; 256 threads -> 2 iterations.
    for (int c = tid; c < MT * (C_IN / 8); c += 256) {
        int row = c >> 4;        // 16 chunks of 8 halves per row
        int ch  = c & 15;
        unsigned long long src =
            (unsigned long long)(featsBf + (size_t)iidx_sh[row] * C_IN + ch * 8);
        // low 32 bits of generic shared pointer == wave-relative LDS offset
        unsigned lds_off = (unsigned)(uintptr_t)(&Ash[row][ch * 8]);
        asm volatile("global_load_async_to_lds_b128 %0, %1, off"
                     :: "v"(lds_off), "v"(src) : "memory");
    }
    asm volatile("s_wait_asynccnt 0x0" ::: "memory");
    __syncthreads();

    const int wave    = tid >> 5;
    const int lane    = tid & 31;
    const int waveRow = wave >> 2;   // 0..1  -> 16-row slab
    const int waveCol = wave & 3;    // 0..3  -> 128-col slab (8 panels)
    const int R       = waveRow * 16;
    const int laneM   = lane & 15;
    const int laneHi  = lane >> 4;

    v8f acc[8] = {};

    const __bf16* wbase =
        wpanel + (size_t)(k * 32 + waveCol * 8) * 16 * C_IN;

    // Pull this wave's 32KB weight-panel slab toward the WGP ahead of use
    // (emits global_prefetch_b8; panels are L2-resident: 3.5MB total).
    __builtin_prefetch(wbase + (size_t)lane * 512, 0, 3);

    #pragma unroll
    for (int kk = 0; kk < C_IN; kk += 32) {
        // A fragment 16x32 bf16: lanes 0-15 hold K kk+0..15,
        // lanes 16-31 hold K kk+16..31 (contiguous, matching B).
        const int kA = kk + laneHi * 16;
        v8bf alo = *(const v8bf*)(&Ash[R + laneM][kA]);
        v8bf ahi = *(const v8bf*)(&Ash[R + laneM][kA + 8]);
        v16bf a = __builtin_shufflevector(alo, ahi,
            0,1,2,3,4,5,6,7,8,9,10,11,12,13,14,15);

        #pragma unroll
        for (int j = 0; j < 8; ++j) {
            // B fragment 32x16 bf16: lane = column, 16 contiguous K per lane
            const __bf16* bp = wbase + ((size_t)j * 16 + laneM) * C_IN
                                     + kk + laneHi * 16;
            v16bf b = *(const v16bf*)bp;
            acc[j] = __builtin_amdgcn_wmma_f32_16x16x32_bf16(
                false, a, false, b, (short)0, acc[j], false, false);
        }
    }

    // Scatter-add C fragments: lane = col (N), VGPR v = row, lanes>=16 -> +8.
    // Accumulator y is 51MB -> L2-resident, so atomics stay on-chip.
    #pragma unroll
    for (int j = 0; j < 8; ++j) {
        const int col = waveCol * 128 + j * 16 + laneM;
        #pragma unroll
        for (int v = 0; v < 8; ++v) {
            int row = R + laneHi * 8 + v;
            int oi  = oidx_sh[row];
            atomicAdd(&y[(size_t)oi * C_OUT + col], acc[j][v]);
        }
    }
}

// ---------------- GeM: clamp -> pow(p) -> segment sums -------------------
__global__ void gem_pow_sum_kernel(const float* __restrict__ y,
                                   const int*   __restrict__ bids,
                                   const float* __restrict__ p,
                                   float* __restrict__ sums,
                                   float* __restrict__ cnt) {
    int idx = blockIdx.x * blockDim.x + threadIdx.x;
    if (idx >= N_OUT * C_OUT) return;
    int row = idx >> 9;      // /512
    int c   = idx & 511;
    int b   = bids[row];
    float v = fmaxf(y[idx], EPS);
    atomicAdd(&sums[b * C_OUT + c], powf(v, p[0]));
    if (c == 0) atomicAdd(&cnt[b], 1.0f);
}

__global__ void gem_final_kernel(const float* __restrict__ sums,
                                 const float* __restrict__ cnt,
                                 const float* __restrict__ p,
                                 float* __restrict__ out) {
    int idx = blockIdx.x * blockDim.x + threadIdx.x;
    if (idx >= NBATCH * C_OUT) return;
    int b = idx >> 9;
    out[idx] = powf(sums[idx] / cnt[b], 1.0f / p[0]);
}

// ---------------- launch --------------------------------------------------
extern "C" void kernel_launch(void* const* d_in, const int* in_sizes, int n_in,
                              void* d_out, int out_size, void* d_ws, size_t ws_size,
                              hipStream_t stream) {
    (void)in_sizes; (void)n_in; (void)out_size; (void)ws_size;

    const float* feats   = (const float*)d_in[0];
    const float* weights = (const float*)d_in[1];
    const float* p       = (const float*)d_in[2];
    const int*   in_idx  = (const int*)d_in[3];
    const int*   out_idx = (const int*)d_in[4];
    const int*   bids    = (const int*)d_in[5];
    float* out = (float*)d_out;

    // workspace layout
    char*  ws       = (char*)d_ws;
    size_t offY     = 0;                                       // 51.2 MB
    size_t offSums  = offY    + (size_t)N_OUT * C_OUT * 4;     // 16 KB
    size_t offCnt   = offSums + (size_t)NBATCH * C_OUT * 4;    // 32 B
    size_t offFeats = (offCnt + NBATCH * 4 + 255) & ~(size_t)255;  // 51.2 MB
    size_t offWp    = offFeats + (size_t)N_IN * C_IN * 2;      // 3.54 MB

    float*  y       = (float*)(ws + offY);
    float*  sums    = (float*)(ws + offSums);
    float*  cnt     = (float*)(ws + offCnt);
    __bf16* featsBf = (__bf16*)(ws + offFeats);
    __bf16* wp      = (__bf16*)(ws + offWp);

    hipMemsetAsync(y, 0, (size_t)N_OUT * C_OUT * 4, stream);
    hipMemsetAsync(sums, 0, ((size_t)NBATCH * C_OUT + NBATCH) * 4, stream);

    {   // feats f32 -> bf16 (vectorized x4)
        int n4 = (N_IN * C_IN) / 4;
        cvt_feats_kernel<<<(n4 + 255) / 256, 256, 0, stream>>>(feats, featsBf, n4);
    }
    {   // weights f32 -> bf16 panel layout
        int tot = KOFF * C_IN * C_OUT;
        cvt_weights_kernel<<<(tot + 255) / 256, 256, 0, stream>>>(weights, wp, tot);
    }

    conv_wmma_kernel<<<KOFF * TILES_PER_K, 256, 0, stream>>>(
        featsBf, wp, in_idx, out_idx, y);

    {
        int tot = N_OUT * C_OUT;
        gem_pow_sum_kernel<<<(tot + 255) / 256, 256, 0, stream>>>(y, bids, p, sums, cnt);
    }
    gem_final_kernel<<<(NBATCH * C_OUT + 255) / 256, 256, 0, stream>>>(sums, cnt, p, out);
}
